// GINNetMultiEdge_54674933678907
// MI455X (gfx1250) — matrix operand order repaired and checked by reading, not compiled, not run on hardware
//
#include <hip/hip_runtime.h>
#include <stdint.h>

// ---------- CDNA5 WMMA types (wave32, gfx1250) ----------
typedef __attribute__((ext_vector_type(16))) __bf16    v16bf;
typedef __attribute__((ext_vector_type(8)))  float     v8f;
typedef __attribute__((ext_vector_type(4)))  float     f4v;   // POD float4 (no ctors -> union/shared safe)
typedef __attribute__((ext_vector_type(4)))  unsigned  u4v;   // POD uint4

union ABfrag { uint32_t u[8]; u4v q[2]; v16bf v; };

__device__ __forceinline__ uint32_t f2bf(float f) {
    uint32_t u = __float_as_uint(f);
    return ((u + 0x7FFFu + ((u >> 16) & 1u)) >> 16) & 0xFFFFu;   // RNE f32 -> bf16
}

__device__ __forceinline__ void loadB(ABfrag& b, const uint32_t* p) {
    b.q[0] = ((const u4v*)p)[0];
    b.q[1] = ((const u4v*)p)[1];
}

__device__ __forceinline__ float dot4(f4v a, f4v c, float bias) {
    return fmaf(a.x, c.x, fmaf(a.y, c.y, fmaf(a.z, c.z, fmaf(a.w, c.w, bias))));
}

// ---------- pack ew2 (f32 [K,K]) into bf16 B-fragment layout [nb][kb][lane][j] ----------
__global__ __launch_bounds__(256) void pack_w2_kernel(const float* __restrict__ w2,
                                                      uint32_t* __restrict__ pk, int Kd) {
    int id = blockIdx.x * 256 + threadIdx.x;
    int total = (Kd * Kd) >> 1;
    if (id >= total) return;
    int KB = Kd >> 5;
    int j    = id & 7;
    int lane = (id >> 3) & 31;
    int t    = id >> 8;
    int kb   = t % KB;
    int nb   = t / KB;
    int n = (nb << 4) + (lane & 15);
    int k = (kb << 5) + (j < 4 ? ((lane >> 4) << 3) + (j << 1) : 16 + ((lane >> 4) << 3) + ((j - 4) << 1));   // B layout: k = hw*8 + 2j (j<4), 16 + hw*8 + 2(j-4) (j>=4)
    uint32_t lo = f2bf(w2[(size_t)k * Kd + n]);
    uint32_t hi = f2bf(w2[(size_t)(k + 1) * Kd + n]);
    pk[id] = lo | (hi << 16);
}

// ---------- transpose ew1 [4,K] -> [K,4] so edge kernel loads one float4 per k ----------
__global__ __launch_bounds__(256) void prep_ew1_kernel(const float* __restrict__ ew1,
                                                       f4v* __restrict__ t, int Kd) {
    int k = blockIdx.x * 256 + threadIdx.x;
    if (k >= Kd) return;
    f4v v; v.x = ew1[k]; v.y = ew1[Kd + k]; v.z = ew1[2 * Kd + k]; v.w = ew1[3 * Kd + k];
    t[k] = v;
}

// ---------- fused NNConv edge kernel ----------
// 32 edges per wave (two 16-row A tiles sharing every B fragment -> 2 WMMAs per
// B b128-pair), double-buffered B loads, occupancy-friendly register budget,
// per-chunk block barrier to keep the 4 waves' identical B streams WGP$-resident.
template <int IN_DIM>
__global__ __launch_bounds__(128, 4) void edge_msg_kernel(
    const float*    __restrict__ x,      // [N, IN_DIM]
    const int*      __restrict__ eidx,   // [2, E]
    int E,
    const f4v*      __restrict__ ew1t,   // [K] transposed ew1
    const float*    __restrict__ eb1,    // [K]
    const float*    __restrict__ ea4,    // [E,4] edge attrs
    const uint32_t* __restrict__ w2pk,   // packed bf16 ew2
    const float*    __restrict__ eb2,    // [K]
    float*          __restrict__ agg,    // [N,32]
    int nTiles)
{
    constexpr int K     = IN_DIM * 32;
    constexpr int KB    = K / 32;     // k-blocks of 32
    constexpr int CHUNK = 4;          // kb per register-resident chunk (K chunked by 128)
    constexpr int NCH   = KB / CHUNK;

    __shared__ float xs  [4][32][32];
    __shared__ float msg [4][32][32];
    __shared__ f4v   eatt[4][32];
    __shared__ int   dsts[4][32];

    const int wv   = threadIdx.x >> 5;
    const int lane = threadIdx.x & 31;
    const int tile = blockIdx.x * 4 + wv;
    const bool active = (tile < nTiles);
    const int e0  = tile * 32;
    const int n15 = lane & 15;
    const int hw  = lane >> 4;

    if (active) {
        int ee = e0 + lane; if (ee >= E) ee = E - 1;
        dsts[wv][lane] = eidx[E + ee];
        eatt[wv][lane] = ((const f4v*)ea4)[ee];
    }
    __syncthreads();

    if (active) {
        #pragma unroll 1
        for (int m = 0; m < 32; ++m) {
            int ee = e0 + m; bool val = (ee < E); if (!val) ee = E - 1;
            int s = eidx[ee];                       // uniform -> scalar load
            float xv = 0.0f;
            if (val && lane < IN_DIM) xv = x[(size_t)s * IN_DIM + lane];
            xs[wv][m][lane] = xv;
        }
    }
    __syncthreads();

    float macc[2][16];
    f4v a40 = {}, a41 = {};
    if (active) {
        #pragma unroll
        for (int r = 0; r < 16; ++r) { macc[0][r] = 0.0f; macc[1][r] = 0.0f; }
        a40 = eatt[wv][n15];        // attrs for A-tile 0 row m=n15
        a41 = eatt[wv][16 + n15];   // attrs for A-tile 1 row m=n15
    }

    #pragma unroll 1
    for (int cc = 0; cc < NCH; ++cc) {
        // keep the block's 4 waves loosely in lockstep so their identical
        // packed-ew2 streams hit in the WGP$ instead of 4x L2 traffic
        __syncthreads();
        if (!active) continue;

        // Build bf16 A fragments (h = relu(ea@ew1+eb1)) for this K-chunk in
        // registers. Each lane computes exactly the (m,k) pairs it consumes
        // (ISA 16-bit A layout); ew1t/eb1 loads shared across both tiles.
        ABfrag h0[CHUNK], h1[CHUNK];
        #pragma unroll
        for (int kb = 0; kb < CHUNK; ++kb) {
            #pragma unroll
            for (int j = 0; j < 8; ++j) {
                int kk = (cc * CHUNK + kb) * 32 +
                         (j < 4 ? hw * 8 + 2 * j : 16 + hw * 8 + 2 * (j - 4));
                f4v  c0 = ew1t[kk],  c1 = ew1t[kk + 1];
                float bb0 = eb1[kk], bb1 = eb1[kk + 1];
                float t00 = dot4(a40, c0, bb0), t01 = dot4(a40, c1, bb1);
                float t10 = dot4(a41, c0, bb0), t11 = dot4(a41, c1, bb1);
                t00 = t00 > 0.0f ? t00 : 0.0f;  t01 = t01 > 0.0f ? t01 : 0.0f;
                t10 = t10 > 0.0f ? t10 : 0.0f;  t11 = t11 > 0.0f ? t11 : 0.0f;
                h0[kb].u[j] = f2bf(t00) | (f2bf(t01) << 16);
                h1[kb].u[j] = f2bf(t10) | (f2bf(t11) << 16);
            }
        }
        const bool last = (cc == NCH - 1);

        #pragma unroll 1
        for (int i = 0; i < IN_DIM; ++i) {
            float xr0[8], xr1[8];
            #pragma unroll
            for (int r = 0; r < 8; ++r) {
                xr0[r] = xs[wv][r + 8 * hw][i];
                xr1[r] = xs[wv][16 + r + 8 * hw][i];
            }
            #pragma unroll
            for (int p = 0; p < 2; ++p) {
                const int nb = 2 * i + p;
                const uint32_t* bp =
                    w2pk + ((size_t)(nb * KB + cc * CHUNK) * 32 + lane) * 8;
                __builtin_prefetch(bp + CHUNK * 256, 0, 1);  // global_prefetch_b8

                // double-buffered B fragments: next kb's loads issue before
                // the WMMAs consuming the current buffer
                ABfrag bb[2];
                loadB(bb[0], bp);
                v8f a0 = {}, a1 = {};
                #pragma unroll
                for (int kb = 0; kb < CHUNK; ++kb) {
                    if (kb + 1 < CHUNK)
                        loadB(bb[(kb + 1) & 1], bp + (size_t)(kb + 1) * 256);
                    a0 = __builtin_amdgcn_wmma_f32_16x16x32_bf16(
                             false, h0[kb].v, false, bb[kb & 1].v,
                             (short)0, a0, false, false);
                    a1 = __builtin_amdgcn_wmma_f32_16x16x32_bf16(
                             false, h1[kb].v, false, bb[kb & 1].v,
                             (short)0, a1, false, false);
                }
                const float b2 = last ? eb2[nb * 16 + n15] : 0.0f;
                #pragma unroll
                for (int r = 0; r < 8; ++r) {
                    macc[0][p * 8 + r] = fmaf(a0[r] + b2, xr0[r], macc[0][p * 8 + r]);
                    macc[1][p * 8 + r] = fmaf(a1[r] + b2, xr1[r], macc[1][p * 8 + r]);
                }
            }
        }
    }

    if (active) {
        #pragma unroll
        for (int r = 0; r < 8; ++r) {
            msg[wv][r + 8 * hw][n15]           = macc[0][r];
            msg[wv][r + 8 * hw][16 + n15]      = macc[0][8 + r];
            msg[wv][16 + r + 8 * hw][n15]      = macc[1][r];
            msg[wv][16 + r + 8 * hw][16 + n15] = macc[1][8 + r];
        }
    }
    __syncthreads();

    if (active) {
        #pragma unroll 1
        for (int m = 0; m < 32; ++m) {
            if (e0 + m < E)
                atomicAdd(&agg[(size_t)dsts[wv][m] * 32 + lane], msg[wv][m][lane]);
        }
    }
}

// ---------- node update: y = agg + x@root + bias; accumulate BN stats ----------
__global__ __launch_bounds__(256) void node_update_kernel(
    const float* __restrict__ xin, int IN_DIM, const float* __restrict__ agg,
    const float* __restrict__ root, const float* __restrict__ bias,
    float* __restrict__ y, float* __restrict__ sums, int N)
{
    __shared__ float sv[256], sq[256];
    int row = blockIdx.x * 8 + (threadIdx.x >> 5);
    int ch  = threadIdx.x & 31;
    float t = 0.0f;
    if (row < N) {
        t = agg[(size_t)row * 32 + ch] + bias[ch];
        for (int i = 0; i < IN_DIM; ++i)
            t = fmaf(xin[(size_t)row * IN_DIM + i], root[i * 32 + ch], t);
        y[(size_t)row * 32 + ch] = t;
    }
    sv[threadIdx.x] = (row < N) ? t : 0.0f;
    sq[threadIdx.x] = (row < N) ? t * t : 0.0f;
    __syncthreads();
    if (threadIdx.x < 32) {
        float s = 0.0f, s2 = 0.0f;
        for (int r = 0; r < 8; ++r) { s += sv[r * 32 + threadIdx.x]; s2 += sq[r * 32 + threadIdx.x]; }
        atomicAdd(&sums[threadIdx.x], s);
        atomicAdd(&sums[32 + threadIdx.x], s2);
    }
}

// ---------- BN + ReLU ----------
__global__ __launch_bounds__(256) void bn_relu_kernel(
    const float* __restrict__ y, const float* __restrict__ sums,
    const float* __restrict__ gamma, const float* __restrict__ beta,
    float* __restrict__ out, int N)
{
    int idx = blockIdx.x * 256 + threadIdx.x;
    if (idx >= N * 32) return;
    int ch = idx & 31;
    float inv = 1.0f / (float)N;
    float mu  = sums[ch] * inv;
    float var = sums[32 + ch] * inv - mu * mu;
    float sc  = gamma[ch] * rsqrtf(var + 1e-5f);
    float sh  = beta[ch] - mu * sc;
    float v   = fmaf(y[idx], sc, sh);
    out[idx]  = v > 0.0f ? v : 0.0f;
}

// ---------- jump concat @ jump_w + pooled segment-sum over 3 graphs ----------
__global__ __launch_bounds__(256) void jump_pool_kernel(
    const float* __restrict__ r0, const float* __restrict__ r1, const float* __restrict__ r2,
    const float* __restrict__ jw, const float* __restrict__ jb,
    const int* __restrict__ batch, float* __restrict__ pooled, int N)
{
    __shared__ float red[256];
    int row = blockIdx.x * 8 + (threadIdx.x >> 5);
    int ch  = threadIdx.x & 31;
    float t = 0.0f; int g = -1;
    if (row < N) {
        t = jb[ch];
        const float* reps[3] = { r0, r1, r2 };
        for (int l = 0; l < 3; ++l)
            for (int i = 0; i < 32; ++i)
                t = fmaf(reps[l][(size_t)row * 32 + i], jw[(l * 32 + i) * 32 + ch], t);
        g = batch[row];
    }
    for (int gg = 0; gg < 3; ++gg) {
        red[threadIdx.x] = (g == gg) ? t : 0.0f;
        __syncthreads();
        if (threadIdx.x < 32) {
            float s = 0.0f;
            for (int r = 0; r < 8; ++r) s += red[r * 32 + threadIdx.x];
            atomicAdd(&pooled[gg * 32 + threadIdx.x], s);
        }
        __syncthreads();
    }
}

// ---------- readout MLP: [3,32] -> relu[3,16] -> [3,1] ----------
__global__ void readout_kernel(const float* __restrict__ pooled,
                               const float* __restrict__ r1w, const float* __restrict__ r1b,
                               const float* __restrict__ r2w, const float* __restrict__ r2b,
                               float* __restrict__ out)
{
    __shared__ float h1[3][16];
    int tid = threadIdx.x;
    if (tid < 48) {
        int g = tid >> 4, j = tid & 15;
        float t = r1b[j];
        for (int i = 0; i < 32; ++i) t = fmaf(pooled[g * 32 + i], r1w[i * 16 + j], t);
        h1[g][j] = t > 0.0f ? t : 0.0f;
    }
    __syncthreads();
    if (tid < 3) {
        float t = r2b[0];
        for (int j = 0; j < 16; ++j) t = fmaf(h1[tid][j], r2w[j], t);
        out[tid] = t;
    }
}

extern "C" void kernel_launch(void* const* d_in, const int* in_sizes, int n_in,
                              void* d_out, int out_size, void* d_ws, size_t ws_size,
                              hipStream_t stream)
{
    const float* x0    = (const float*)d_in[0];
    const int*   eidx  = (const int*)  d_in[1];
    const float* ea    = (const float*)d_in[2];
    const int*   batch = (const int*)  d_in[3];
    const int N = in_sizes[0] / 16;
    const int E = in_sizes[1] / 2;

    // workspace carve-out (256B aligned)
    char* ws = (char*)d_ws;
    auto carve = [&](size_t bytes) { char* p = ws; ws += (bytes + 255) & ~(size_t)255; return p; };
    float*    rep[3];
    for (int l = 0; l < 3; ++l) rep[l] = (float*)carve((size_t)N * 32 * 4);
    float*    agg    = (float*)   carve((size_t)N * 32 * 4);
    float*    y      = (float*)   carve((size_t)N * 32 * 4);
    float*    sums   = (float*)   carve(64 * 4);
    float*    pooled = (float*)   carve(96 * 4);
    f4v*      ew1t   = (f4v*)     carve(1024 * 16);
    uint32_t* w2pk   = (uint32_t*)carve((size_t)1024 * 1024 / 2 * 4);

    hipMemsetAsync(pooled, 0, 96 * 4, stream);

    const float* xin = x0;
    const int nTiles  = (E + 31) / 32;
    const int nBlocks = (nTiles + 3) / 4;

    for (int l = 0; l < 3; ++l) {
        const int IN_DIM = (l == 0) ? 16 : 32;
        const int K = IN_DIM * 32;
        const int base = 4 + l * 8;
        const float* ew1   = (const float*)d_in[base + 0];
        const float* eb1   = (const float*)d_in[base + 1];
        const float* ew2   = (const float*)d_in[base + 2];
        const float* eb2   = (const float*)d_in[base + 3];
        const float* root  = (const float*)d_in[base + 4];
        const float* bias  = (const float*)d_in[base + 5];
        const float* gamma = (const float*)d_in[base + 6];
        const float* beta  = (const float*)d_in[base + 7];

        hipMemsetAsync(agg, 0, (size_t)N * 32 * 4, stream);
        hipMemsetAsync(sums, 0, 64 * 4, stream);
        prep_ew1_kernel<<<(K + 255) / 256, 256, 0, stream>>>(ew1, ew1t, K);
        int totalPk = K * K / 2;
        pack_w2_kernel<<<(totalPk + 255) / 256, 256, 0, stream>>>(ew2, w2pk, K);

        if (IN_DIM == 16)
            edge_msg_kernel<16><<<nBlocks, 128, 0, stream>>>(
                xin, eidx, E, ew1t, eb1, ea, w2pk, eb2, agg, nTiles);
        else
            edge_msg_kernel<32><<<nBlocks, 128, 0, stream>>>(
                xin, eidx, E, ew1t, eb1, ea, w2pk, eb2, agg, nTiles);

        node_update_kernel<<<(N + 7) / 8, 256, 0, stream>>>(xin, IN_DIM, agg, root, bias, y, sums, N);
        bn_relu_kernel<<<(N * 32 + 255) / 256, 256, 0, stream>>>(y, sums, gamma, beta, rep[l], N);
        xin = rep[l];
    }

    jump_pool_kernel<<<(N + 7) / 8, 256, 0, stream>>>(
        rep[0], rep[1], rep[2], (const float*)d_in[28], (const float*)d_in[29],
        batch, pooled, N);
    readout_kernel<<<1, 64, 0, stream>>>(
        pooled, (const float*)d_in[30], (const float*)d_in[31],
        (const float*)d_in[32], (const float*)d_in[33], (float*)d_out);
}
